// HistoryModelBlind_78460462563899
// MI455X (gfx1250) — matrix operand order repaired and checked by reading, not compiled, not run on hardware
//
#include <hip/hip_runtime.h>
#include <hip/hip_bf16.h>
#include <math.h>

// ---------------- problem constants (from reference) ----------------
constexpr int B_ = 64, N_ = 6, S_ = 256, T_ = 64;
constexpr int D_ = 512, FF_ = 512, H_ = 4, L_ = 4, IMG_ = 2048, DH_ = 128;

// ---------------- WMMA types ----------------
typedef __attribute__((ext_vector_type(16))) __bf16 v16bf;
typedef __attribute__((ext_vector_type(8)))  float  v8f;

// =====================================================================
// GEMM: C[M,N] = act(A[M,K] @ W[K,N] + bias[N])   (fp32 -> bf16 WMMA)
// Block tile 128x128, K step 32.  256 threads = 8 wave32 in 4(M)x2(N);
// each wave owns a 32x64 patch = 2x4 v_wmma_f32_16x16x32_bf16 tiles
// (8 WMMA per K step).  Staging is float4 global loads -> packed bf16
// -> b64 LDS stores, software-pipelined through registers.
// Requires: M%128==0, N%128==0, K%32==0 (true for all shapes here).
// =====================================================================
__global__ __launch_bounds__(256) void gemm_bias_kernel(
    const float* __restrict__ A, const float* __restrict__ W,
    const float* __restrict__ bias, float* __restrict__ C,
    int M, int N, int K, int do_relu)
{
    constexpr int BM = 128, BN = 128, BK = 32;
    constexpr int ASTR = BK + 4;    // 36 elems (72 B rows, 8B aligned)
    constexpr int BSTR = BN + 8;    // 136 elems (272 B rows, 8B aligned)
    __shared__ __bf16 As[BM][ASTR];
    __shared__ __bf16 Bs[BK][BSTR];

    const int tid  = threadIdx.x;
    const int wave = tid >> 5;
    const int lane = tid & 31;
    const int wm   = wave >> 1;          // 0..3 -> rows wm*32
    const int wn   = wave & 1;           // 0..1 -> cols wn*64
    const int rowBase = blockIdx.y * BM;
    const int colBase = blockIdx.x * BN;

    // ---- per-thread staging addresses (computed once) ----
    const int arow = tid >> 3;           // 0..31  (+ i*32)
    const int acol = (tid & 7) * 4;      // 0..28
    const int brow = tid >> 5;           // 0..7   (+ i*8)
    const int bcol = (tid & 31) * 4;     // 0..124
    const float* aG = A + (size_t)(rowBase + arow) * K + acol;
    const float* bG = W + (size_t)brow * N + colBase + bcol;
    __bf16* aL = &As[arow][acol];
    __bf16* bL = &Bs[brow][bcol];

    float4 ra[4], rb[4];
    auto loadT = [&](int k0) {
        #pragma unroll
        for (int i = 0; i < 4; ++i)
            ra[i] = *(const float4*)(aG + (size_t)i * 32 * K + k0);
        #pragma unroll
        for (int i = 0; i < 4; ++i)
            rb[i] = *(const float4*)(bG + (size_t)(i * 8 + k0) * N);
    };
    auto storeT = [&]() {
        #pragma unroll
        for (int i = 0; i < 4; ++i) {
            union { __bf16 h[4]; unsigned long long u; } t;
            t.h[0] = (__bf16)ra[i].x; t.h[1] = (__bf16)ra[i].y;
            t.h[2] = (__bf16)ra[i].z; t.h[3] = (__bf16)ra[i].w;
            *(unsigned long long*)(aL + i * 32 * ASTR) = t.u;
        }
        #pragma unroll
        for (int i = 0; i < 4; ++i) {
            union { __bf16 h[4]; unsigned long long u; } t;
            t.h[0] = (__bf16)rb[i].x; t.h[1] = (__bf16)rb[i].y;
            t.h[2] = (__bf16)rb[i].z; t.h[3] = (__bf16)rb[i].w;
            *(unsigned long long*)(bL + i * 8 * BSTR) = t.u;
        }
    };

    v8f acc[2][4] = {};
    const int mrow  = lane & 15;
    const int khalf = (lane >> 4) * 8;

    loadT(0);
    for (int k0 = 0; k0 < K; k0 += BK) {
        storeT();                         // regs -> LDS (bf16)
        __syncthreads();
        if (k0 + BK < K) loadT(k0 + BK);  // pipeline next tile into regs
        if (k0 + 2 * BK < K)              // gfx1250 global_prefetch_b8
            __builtin_prefetch(aG + (size_t)k0 + 2 * BK, 0, 1);

        // ---- build fragments per ISA lane layout ----
        union Frag { v16bf v; unsigned u[8]; };
        Frag a[2], b[4];
        #pragma unroll
        for (int t = 0; t < 2; ++t) {
            const int am = wm * 32 + t * 16 + mrow;    // A: lane owns row am
            #pragma unroll
            for (int j = 0; j < 8; ++j) {              // k = (e/8)*16+khalf+e%8
                int k = (j >> 2) * 16 + khalf + (j & 3) * 2;
                a[t].u[j] = *(const unsigned*)&As[am][k];
            }
        }
        #pragma unroll
        for (int t = 0; t < 4; ++t) {
            const int bn = wn * 64 + t * 16 + mrow;    // B: lane owns col bn, elems = k
            #pragma unroll
            for (int e = 0; e < 16; ++e)
                b[t].v[e] = Bs[(e >> 3) * 16 + khalf + (e & 7)][bn];
        }

        #pragma unroll
        for (int tm = 0; tm < 2; ++tm)
            #pragma unroll
            for (int tn = 0; tn < 4; ++tn)
                acc[tm][tn] = __builtin_amdgcn_wmma_f32_16x16x32_bf16(
                    false, a[tm].v, false, b[tn].v,
                    (short)0, acc[tm][tn], false, false);

        __syncthreads();
    }

    // ---- store: C/D layout row = r + (lane/16)*8, col = lane%16 ----
    #pragma unroll
    for (int tm = 0; tm < 2; ++tm)
        #pragma unroll
        for (int tn = 0; tn < 4; ++tn) {
            const int col = colBase + wn * 64 + tn * 16 + (lane & 15);
            const float bv = bias[col];
            #pragma unroll
            for (int r = 0; r < 8; ++r) {
                const int row = rowBase + wm * 32 + tm * 16 + r + (lane >> 4) * 8;
                float v = acc[tm][tn][r] + bv;
                if (do_relu) v = fmaxf(v, 0.f);
                C[(size_t)row * N + col] = v;
            }
        }
}

// =====================================================================
// Embedding gather + sinusoidal positional encoding (computed on the fly)
// =====================================================================
__global__ void embed_pe_kernel(const int* __restrict__ tok,
                                const float* __restrict__ emb,
                                float* __restrict__ x, int rows, int seqlen)
{
    int idx = blockIdx.x * blockDim.x + threadIdx.x;
    if (idx >= rows * D_) return;
    int r = idx >> 9;            // / 512
    int d = idx & 511;
    int pos = r % seqlen;
    int t = tok[r];
    float freq = __expf(-(float)(d & ~1) * (9.210340371976184f / 512.0f)); // ln(1e4)/D
    float ang  = (float)pos * freq;
    float pe   = (d & 1) ? __cosf(ang) : __sinf(ang);
    x[idx] = emb[(size_t)t * D_ + d] + pe;
}

// =====================================================================
// Attention: one block per (sequence m, head h).  DH = 128 = blockDim.
// scores -> masked softmax -> ctx.  K/V rows hit L2 across q iterations.
// =====================================================================
__global__ __launch_bounds__(128) void attention_kernel(
    const float* __restrict__ Qm, const float* __restrict__ Km,
    const float* __restrict__ Vm, float* __restrict__ Om,
    const int* __restrict__ lens, int Tt)
{
    extern __shared__ float smem[];
    float* scores = smem;            // Tt (multiple of 64 -> qrow stays aligned)
    float* qrow   = smem + Tt;       // 128
    float* red    = qrow + DH_;      // 128

    const int m = blockIdx.x, h = blockIdx.y, tid = threadIdx.x;
    int valid = Tt;
    if (lens) { int l = lens[m]; valid = (l < 1) ? 1 : l; }
    const float scale = 0.08838834764831845f;  // 1/sqrt(128)

    for (int q = 0; q < Tt; ++q) {
        qrow[tid] = Qm[((size_t)(m * Tt + q)) * D_ + h * DH_ + tid];
        __syncthreads();

        for (int k = tid; k < Tt; k += 128) {
            const float4* kr = (const float4*)&Km[((size_t)(m * Tt + k)) * D_ + h * DH_];
            const float4* qr = (const float4*)qrow;
            float s = 0.f;
            #pragma unroll
            for (int d = 0; d < DH_ / 4; ++d) {
                float4 kv = kr[d], qv = qr[d];
                s += qv.x * kv.x + qv.y * kv.y + qv.z * kv.z + qv.w * kv.w;
            }
            s *= scale;
            if (k >= valid) s = -1e9f;
            scores[k] = s;
        }
        __syncthreads();

        float lm = -3.4e38f;
        for (int k = tid; k < Tt; k += 128) lm = fmaxf(lm, scores[k]);
        red[tid] = lm; __syncthreads();
        for (int s2 = 64; s2 > 0; s2 >>= 1) {
            if (tid < s2) red[tid] = fmaxf(red[tid], red[tid + s2]);
            __syncthreads();
        }
        const float mx = red[0];
        __syncthreads();

        float ls = 0.f;
        for (int k = tid; k < Tt; k += 128) {
            float e = __expf(scores[k] - mx);
            scores[k] = e; ls += e;
        }
        red[tid] = ls; __syncthreads();
        for (int s2 = 64; s2 > 0; s2 >>= 1) {
            if (tid < s2) red[tid] += red[tid + s2];
            __syncthreads();
        }
        const float inv = 1.f / red[0];

        float o = 0.f;
        #pragma unroll 4
        for (int k = 0; k < Tt; ++k)
            o += scores[k] * Vm[((size_t)(m * Tt + k)) * D_ + h * DH_ + tid];
        Om[((size_t)(m * Tt + q)) * D_ + h * DH_ + tid] = o * inv;
        __syncthreads();
    }
}

// =====================================================================
// Fused residual add + LayerNorm over rows of 512 (in place on x)
// =====================================================================
__global__ __launch_bounds__(256) void add_ln_kernel(
    float* __restrict__ x, const float* __restrict__ delta,
    const float* __restrict__ g, const float* __restrict__ b)
{
    __shared__ float red[256];
    const size_t r = blockIdx.x;
    const int tid = threadIdx.x;
    float v0 = x[r * D_ + tid]       + delta[r * D_ + tid];
    float v1 = x[r * D_ + tid + 256] + delta[r * D_ + tid + 256];

    red[tid] = v0 + v1; __syncthreads();
    for (int t = 128; t > 0; t >>= 1) {
        if (tid < t) red[tid] += red[tid + t];
        __syncthreads();
    }
    const float mu = red[0] * (1.f / 512.f);
    __syncthreads();

    const float d0 = v0 - mu, d1 = v1 - mu;
    red[tid] = d0 * d0 + d1 * d1; __syncthreads();
    for (int t = 128; t > 0; t >>= 1) {
        if (tid < t) red[tid] += red[tid + t];
        __syncthreads();
    }
    const float inv = rsqrtf(red[0] * (1.f / 512.f) + 1e-5f);

    x[r * D_ + tid]       = d0 * inv * g[tid]       + b[tid];
    x[r * D_ + tid + 256] = d1 * inv * g[tid + 256] + b[tid + 256];
}

// =====================================================================
// Final: sep += last_token(hout); optional relu+L2 normalize; dot with
// batch_hidden (= encoder1 out at s = S-1).  One block per (b,n).
// =====================================================================
__global__ __launch_bounds__(128) void final_kernel(
    const float* __restrict__ sep, const float* __restrict__ x2,
    const float* __restrict__ x1,  const int* __restrict__ lens,
    const int* __restrict__ normalize, float* __restrict__ out)
{
    __shared__ float vbuf[D_];
    __shared__ float red[128];
    const int bn = blockIdx.x;
    const int b  = bn / N_;
    const int tid = threadIdx.x;
    const int len = lens[bn];
    const int idx = (len > 0) ? (len - 1) : 0;
    const int norm = normalize[0];

    float nrm = 0.f;
    for (int d = tid; d < D_; d += 128) {
        float last = (len > 0) ? x2[((size_t)(bn * T_ + idx)) * D_ + d] : 0.f;
        float v = sep[(size_t)bn * D_ + d] + last;
        if (norm) { v = fmaxf(v, 0.f); nrm += v * v; }
        vbuf[d] = v;
    }
    red[tid] = nrm; __syncthreads();
    for (int t = 64; t > 0; t >>= 1) {
        if (tid < t) red[tid] += red[tid + t];
        __syncthreads();
    }
    float scale = 1.f;
    if (norm) scale = 1.f / fmaxf(sqrtf(red[0]), 1e-12f);
    __syncthreads();

    float dot = 0.f;
    for (int d = tid; d < D_; d += 128)
        dot += vbuf[d] * scale * x1[((size_t)(b * S_ + (S_ - 1))) * D_ + d];
    red[tid] = dot; __syncthreads();
    for (int t = 64; t > 0; t >>= 1) {
        if (tid < t) red[tid] += red[tid + t];
        __syncthreads();
    }
    if (tid == 0) out[bn] = red[0];
}

// =====================================================================
// Host orchestration
// =====================================================================
extern "C" void kernel_launch(void* const* d_in, const int* in_sizes, int n_in,
                              void* d_out, int out_size, void* d_ws, size_t ws_size,
                              hipStream_t stream)
{
    const int*   segment      = (const int*)  d_in[0];
    const int*   prev_hist    = (const int*)  d_in[1];
    const int*   hist_lengths = (const int*)  d_in[2];
    const float* sep_img      = (const float*)d_in[3];
    const int*   normalize    = (const int*)  d_in[4];
    const float* emb          = (const float*)d_in[5];
    const float* Wq  = (const float*)d_in[6];  const float* bq  = (const float*)d_in[7];
    const float* Wk  = (const float*)d_in[8];  const float* bk  = (const float*)d_in[9];
    const float* Wv  = (const float*)d_in[10]; const float* bv  = (const float*)d_in[11];
    const float* Wo  = (const float*)d_in[12]; const float* bo  = (const float*)d_in[13];
    const float* ln1g = (const float*)d_in[14]; const float* ln1b = (const float*)d_in[15];
    const float* W1  = (const float*)d_in[16]; const float* b1  = (const float*)d_in[17];
    const float* W2  = (const float*)d_in[18]; const float* b2  = (const float*)d_in[19];
    const float* ln2g = (const float*)d_in[20]; const float* ln2b = (const float*)d_in[21];
    const float* Wsep = (const float*)d_in[22]; const float* bsep = (const float*)d_in[23];
    float* out = (float*)d_out;

    // workspace carve-up (floats): x1 + x2 + 4 activation scratch + sep
    float* ws = (float*)d_ws;
    const size_t T1 = (size_t)B_ * S_ * D_;        // 8.39 M
    const size_t T2 = (size_t)B_ * N_ * T_ * D_;   // 12.58 M (max token count)
    float* x1   = ws;
    float* x2   = x1   + T1;
    float* bufA = x2   + T2;
    float* bufB = bufA + T2;
    float* bufC = bufB + T2;
    float* bufD = bufC + T2;
    float* sep  = bufD + T2;                       // 384*512

    auto gemm = [&](const float* Ap, const float* Wp, const float* biasp, float* Cp,
                    int M, int Nn, int Kk, int relu) {
        dim3 g(Nn / 128, M / 128);
        gemm_bias_kernel<<<g, 256, 0, stream>>>(Ap, Wp, biasp, Cp, M, Nn, Kk, relu);
    };

    auto encoder = [&](float* x, int Mseq, int Tt, const int* lensp) {
        const int Mtok = Mseq * Tt;
        const size_t smem = (size_t)(Tt + 2 * DH_) * sizeof(float);
        for (int l = 0; l < L_; ++l) {
            gemm(x, Wq + (size_t)l * D_ * D_, bq + l * D_, bufA, Mtok, D_, D_, 0);
            gemm(x, Wk + (size_t)l * D_ * D_, bk + l * D_, bufB, Mtok, D_, D_, 0);
            gemm(x, Wv + (size_t)l * D_ * D_, bv + l * D_, bufC, Mtok, D_, D_, 0);
            attention_kernel<<<dim3(Mseq, H_), 128, smem, stream>>>(
                bufA, bufB, bufC, bufD, lensp, Tt);
            gemm(bufD, Wo + (size_t)l * D_ * D_, bo + l * D_, bufA, Mtok, D_, D_, 0);
            add_ln_kernel<<<Mtok, 256, 0, stream>>>(x, bufA, ln1g + l * D_, ln1b + l * D_);
            gemm(x, W1 + (size_t)l * D_ * FF_, b1 + l * FF_, bufB, Mtok, FF_, D_, 1);
            gemm(bufB, W2 + (size_t)l * FF_ * D_, b2 + l * D_, bufC, Mtok, D_, FF_, 0);
            add_ln_kernel<<<Mtok, 256, 0, stream>>>(x, bufC, ln2g + l * D_, ln2b + l * D_);
        }
    };

    // ---- encoder 1: segment batch (64 seqs x 256 tokens), no mask ----
    {
        int rows = B_ * S_, total = rows * D_;
        embed_pe_kernel<<<(total + 255) / 256, 256, 0, stream>>>(segment, emb, x1, rows, S_);
    }
    encoder(x1, B_, S_, nullptr);

    // ---- image projection: (384 x 2048) @ (2048 x 512) ----
    gemm(sep_img, Wsep, bsep, sep, B_ * N_, FF_, IMG_, 0);

    // ---- encoder 2: histories (384 seqs x 64 tokens), length mask ----
    {
        int rows = B_ * N_ * T_, total = rows * D_;
        embed_pe_kernel<<<(total + 255) / 256, 256, 0, stream>>>(prev_hist, emb, x2, rows, T_);
    }
    encoder(x2, B_ * N_, T_, hist_lengths);

    // ---- last-token gather + relu/normalize + dot ----
    final_kernel<<<B_ * N_, 128, 0, stream>>>(sep, x2, x1, hist_lengths, normalize, out);
}
